// build_model_23716809409142
// MI455X (gfx1250) — matrix operand order repaired and checked by reading, not compiled, not run on hardware
//
#include <hip/hip_runtime.h>
#include <hip/hip_bf16.h>

typedef __attribute__((ext_vector_type(8)))  float    v8f;
typedef __attribute__((ext_vector_type(8)))  _Float16 v8h;
typedef __attribute__((ext_vector_type(16))) _Float16 v16h;

static constexpr int Bc = 64, Tc = 512, Ic = 512, Hc = 1024, G3 = 3072;
static constexpr int NWG = 32;   // persistent workgroups for the scan kernel

#define WMMA_F16(a, b, c) \
  __builtin_amdgcn_wmma_f32_16x16x32_f16(false, (a), false, (b), (short)0, (c), false, false)

__device__ __forceinline__ float sigf(float x) { return 1.0f / (1.0f + __expf(-x)); }

// ---- WMMA fragment loaders ------------------------------------------------
// A is row-major [M,K] f16, computing X @ W^T with W row-major [N,K].
// A 16x32 fragment, contiguous k order used here (same as B): lane L (row M=L%16):
//   L<16 : elems 0..15 = K k0+0..15
//   L>=16: elems 0..15 = K k0+16..31
__device__ __forceinline__ v16h load_a_frag(const _Float16* __restrict__ A,
                                            long row0, int K, int k0, int lane) {
  const _Float16* p = A + (row0 + (lane & 15)) * (long)K + k0 + ((lane & 16) ? 16 : 0);
  v16h out;
  *reinterpret_cast<v8h*>(&out)       = *reinterpret_cast<const v8h*>(p);
  *(reinterpret_cast<v8h*>(&out) + 1) = *reinterpret_cast<const v8h*>(p + 8);
  return out;
}
// B 32x16 fragment, same contiguous k order as A: lane L (col N=L%16) holds 16 consecutive K values
// starting at k0 (lanes 0-15) or k0+16 (lanes 16-31) -> one contiguous 32B load
// from row (col) of row-major W[N,K].
__device__ __forceinline__ v16h load_b_frag(const _Float16* __restrict__ W,
                                            long row0, int K, int k0, int lane) {
  const _Float16* p = W + (row0 + (lane & 15)) * (long)K + k0 + ((lane & 16) ? 16 : 0);
  return *reinterpret_cast<const v16h*>(p);
}
// D/C: v8f element r at lane L = C[r + (L>=16?8:0)][L%16]

// ---- device-wide barrier for the persistent scan kernel -------------------
__device__ __forceinline__ void grid_sync(unsigned* bar, unsigned target) {
  __threadfence();
  __syncthreads();
  if (threadIdx.x == 0) {
    atomicAdd(bar, 1u);
    while (__hip_atomic_load(bar, __ATOMIC_ACQUIRE, __HIP_MEMORY_SCOPE_AGENT) < target)
      __builtin_amdgcn_s_sleep(1);
  }
  __syncthreads();
}

// ---- small utility kernels ------------------------------------------------
__global__ __launch_bounds__(256) void k_cvt(const float* __restrict__ s,
                                             _Float16* __restrict__ d, long n) {
  long i = (long)blockIdx.x * blockDim.x + threadIdx.x;
  long stride = (long)gridDim.x * blockDim.x;
  for (; i < n; i += stride) d[i] = (_Float16)s[i];
}

// pad w_r [1,H] into [16,H] f16 (rows 1..15 zero) so the r-head is a WMMA tile
__global__ __launch_bounds__(256) void k_pad_wr(const float* __restrict__ w_r,
                                                _Float16* __restrict__ d) {
  int i = blockIdx.x * 256 + threadIdx.x;            // 16*1024 threads
  if (i < 16 * Hc) d[i] = (i < Hc) ? (_Float16)w_r[i] : (_Float16)0.0f;
}

__global__ __launch_bounds__(256) void k_init(const float* __restrict__ s,
                                              _Float16* __restrict__ sp16,
                                              float* __restrict__ spf,
                                              unsigned* __restrict__ bar) {
  int i = blockIdx.x * 256 + threadIdx.x;            // 64*1024 threads
  if (i == 0) *bar = 0u;                             // reset barrier each launch
  if (i < Bc * Hc) {
    float v = s[i];
    spf[i]  = v;
    sp16[i] = (_Float16)v;
  }
}

// ---- Kernel 1: fused layer-0 GEMM + gates -> h0 (f16) ---------------------
// h0 = (1-z)*n ; r = sig(xr + b_hh0_r); z = sig(xz + b_hh0_z); n = tanh(xn + r*b_hh0_n)
__global__ __launch_bounds__(256) void k_layer0(const _Float16* __restrict__ a16,
                                                const _Float16* __restrict__ wih0,
                                                const float* __restrict__ b_ih0,
                                                const float* __restrict__ b_hh0,
                                                _Float16* __restrict__ h0_16) {
  const int lane = threadIdx.x & 31;
  const int wv   = threadIdx.x >> 5;
  const int tile = blockIdx.x * 8 + wv;              // 2048*64 tiles
  const int mt = tile >> 6, nt = tile & 63;
  const long rowBase = (long)mt * 16;
  const int  hb = nt * 16;
  v8f ar = {}, az = {}, an = {};
  for (int k0 = 0; k0 < Ic; k0 += 32) {
    v16h af = load_a_frag(a16, rowBase, Ic, k0, lane);
    v16h br = load_b_frag(wih0, 0 * Hc + hb, Ic, k0, lane);
    v16h bz = load_b_frag(wih0, 1 * Hc + hb, Ic, k0, lane);
    v16h bn = load_b_frag(wih0, 2 * Hc + hb, Ic, k0, lane);
    ar = WMMA_F16(af, br, ar);
    az = WMMA_F16(af, bz, az);
    an = WMMA_F16(af, bn, an);
  }
  const int col = lane & 15, rofs = (lane & 16) ? 8 : 0;
  const int h = hb + col;
  const float bir = b_ih0[h], biz = b_ih0[Hc + h], bin_ = b_ih0[2 * Hc + h];
  const float bhr = b_hh0[h], bhz = b_hh0[Hc + h], bhn = b_hh0[2 * Hc + h];
  for (int r = 0; r < 8; ++r) {
    long row = rowBase + r + rofs;
    float rg = sigf(ar[r] + bir + bhr);
    float zg = sigf(az[r] + biz + bhz);
    float ng = tanhf(an[r] + bin_ + rg * bhn);
    h0_16[row * Hc + h] = (_Float16)((1.0f - zg) * ng);
  }
}

// ---- Kernel 2: xp1 = h0 @ w_ih1^T + b_ih1 (f16 out) -----------------------
__global__ __launch_bounds__(256) void k_xp1(const _Float16* __restrict__ h0_16,
                                             const _Float16* __restrict__ wih1,
                                             const float* __restrict__ b_ih1,
                                             _Float16* __restrict__ xp1) {
  const int lane = threadIdx.x & 31;
  const int wv   = threadIdx.x >> 5;
  const int tile = blockIdx.x * 8 + wv;              // 2048*192 tiles
  const int mt = tile / 192, nt = tile % 192;
  const long rowBase = (long)mt * 16;
  const int  jb = nt * 16;
  v8f acc = {};
  for (int k0 = 0; k0 < Hc; k0 += 32) {
    v16h af = load_a_frag(h0_16, rowBase, Hc, k0, lane);
    v16h bf = load_b_frag(wih1, jb, Hc, k0, lane);
    acc = WMMA_F16(af, bf, acc);
  }
  const int col = lane & 15, rofs = (lane & 16) ? 8 : 0;
  const int j = jb + col;
  const float bias = b_ih1[j];
  for (int r = 0; r < 8; ++r)
    xp1[(rowBase + r + rofs) * (long)G3 + j] = (_Float16)(acc[r] + bias);
}

// ---- Kernel 3: persistent cooperative scan over T -------------------------
__global__ __launch_bounds__(256) void k_scan(
    const _Float16* __restrict__ xp1, const _Float16* __restrict__ whh1,
    const float* __restrict__ b_hh1,
    const _Float16* __restrict__ ws16, const float* __restrict__ b_s,
    const _Float16* __restrict__ wrp, const float* __restrict__ b_r,
    _Float16* __restrict__ sp16, float* __restrict__ spf,
    _Float16* __restrict__ h1_16,
    float* __restrict__ r_out, float* __restrict__ s_out,
    unsigned* __restrict__ bar) {
  const int lane = threadIdx.x & 31;
  const int wv   = threadIdx.x >> 5;
  const int gw   = blockIdx.x * 8 + wv;              // 0..255, one tile per wave
  const int col  = lane & 15, rofs = (lane & 16) ? 8 : 0;
  const int mt = gw >> 6, nt = gw & 63;
  const int hb = nt * 16, h = hb + col;
  const float bhr = b_hh1[h], bhz = b_hh1[Hc + h], bhn = b_hh1[2 * Hc + h];
  const float bsv = b_s[h], brv = b_r[0];
  unsigned phase = 0;

  for (int t = 0; t < Tc; ++t) {
    // -------- Phase A: hp1 GEMM + GRU gates -> h1 --------
    v8f accr = {}, accz = {}, accn = {};
    for (int k0 = 0; k0 < Hc; k0 += 32) {
      v16h af = load_a_frag(sp16, (long)mt * 16, Hc, k0, lane);
      v16h br = load_b_frag(whh1, 0 * Hc + hb, Hc, k0, lane);
      v16h bz = load_b_frag(whh1, 1 * Hc + hb, Hc, k0, lane);
      v16h bn = load_b_frag(whh1, 2 * Hc + hb, Hc, k0, lane);
      accr = WMMA_F16(af, br, accr);
      accz = WMMA_F16(af, bz, accz);
      accn = WMMA_F16(af, bn, accn);
    }
    for (int r = 0; r < 8; ++r) {
      int b = mt * 16 + r + rofs;
      long xrow = ((long)b * Tc + t) * G3;
      float xr = (float)xp1[xrow + h];
      float xz = (float)xp1[xrow + Hc + h];
      float xn = (float)xp1[xrow + 2 * Hc + h];
      float rg = sigf(xr + accr[r] + bhr);
      float zg = sigf(xz + accz[r] + bhz);
      float ng = tanhf(xn + rg * (accn[r] + bhn));
      float h1 = (1.0f - zg) * ng + zg * spf[b * Hc + h];
      h1_16[b * Hc + h] = (_Float16)h1;
    }
    ++phase;
    grid_sync(bar, phase * NWG);

    // -------- Phase B: s head (and r head on waves 0..3) --------
    {
      v8f acc = {};
      for (int k0 = 0; k0 < Hc; k0 += 32) {
        v16h af = load_a_frag(h1_16, (long)mt * 16, Hc, k0, lane);
        v16h bf = load_b_frag(ws16, hb, Hc, k0, lane);
        acc = WMMA_F16(af, bf, acc);
      }
      for (int r = 0; r < 8; ++r) {
        int b = mt * 16 + r + rofs;
        float v = sigf(acc[r] + bsv);
        s_out[((long)b * Tc + t) * Hc + h] = v;
        spf[b * Hc + h]  = v;
        sp16[b * Hc + h] = (_Float16)v;
      }
    }
    if (gw < 4) {                                    // r head: padded [16,H] weight
      v8f acc = {};
      for (int k0 = 0; k0 < Hc; k0 += 32) {
        v16h af = load_a_frag(h1_16, (long)gw * 16, Hc, k0, lane);
        v16h bf = load_b_frag(wrp, 0, Hc, k0, lane);
        acc = WMMA_F16(af, bf, acc);
      }
      if (col == 0) {
        for (int r = 0; r < 8; ++r) {
          int b = gw * 16 + r + rofs;
          r_out[(long)b * Tc + t] = sigf(acc[r] + brv);
        }
      }
    }
    ++phase;
    grid_sync(bar, phase * NWG);
  }
}

// ---- host launch ----------------------------------------------------------
extern "C" void kernel_launch(void* const* d_in, const int* in_sizes, int n_in,
                              void* d_out, int out_size, void* d_ws, size_t ws_size,
                              hipStream_t stream) {
  (void)in_sizes; (void)n_in; (void)out_size; (void)ws_size;
  const float* s     = (const float*)d_in[0];
  const float* a     = (const float*)d_in[1];
  const float* w_ih0 = (const float*)d_in[2];
  /* d_in[3] = w_hh0: unused (layer-0 hidden is zero every step) */
  const float* b_ih0 = (const float*)d_in[4];
  const float* b_hh0 = (const float*)d_in[5];
  const float* w_ih1 = (const float*)d_in[6];
  const float* w_hh1 = (const float*)d_in[7];
  const float* b_ih1 = (const float*)d_in[8];
  const float* b_hh1 = (const float*)d_in[9];
  const float* w_r   = (const float*)d_in[10];
  const float* b_r   = (const float*)d_in[11];
  const float* w_s   = (const float*)d_in[12];
  const float* b_s   = (const float*)d_in[13];

  size_t off = 0;
  auto take = [&](size_t bytes) -> void* {
    void* q = (char*)d_ws + off;
    off += (bytes + 255) & ~(size_t)255;
    return q;
  };
  _Float16* a16     = (_Float16*)take((size_t)Bc * Tc * Ic * 2);
  _Float16* wih0_16 = (_Float16*)take((size_t)G3 * Ic * 2);
  _Float16* wih1_16 = (_Float16*)take((size_t)G3 * Hc * 2);
  _Float16* whh1_16 = (_Float16*)take((size_t)G3 * Hc * 2);
  _Float16* ws16    = (_Float16*)take((size_t)Hc * Hc * 2);
  _Float16* wrp16   = (_Float16*)take((size_t)16 * Hc * 2);
  _Float16* h0_16   = (_Float16*)take((size_t)Bc * Tc * Hc * 2);
  _Float16* xp1_16  = (_Float16*)take((size_t)Bc * Tc * G3 * 2);
  _Float16* sp16    = (_Float16*)take((size_t)Bc * Hc * 2);
  float*    spf     = (float*)take((size_t)Bc * Hc * 4);
  _Float16* h1_16   = (_Float16*)take((size_t)Bc * Hc * 2);
  unsigned* bar     = (unsigned*)take(256);

  k_cvt<<<2048, 256, 0, stream>>>(a, a16, (long)Bc * Tc * Ic);
  k_cvt<<<512, 256, 0, stream>>>(w_ih0, wih0_16, (long)G3 * Ic);
  k_cvt<<<512, 256, 0, stream>>>(w_ih1, wih1_16, (long)G3 * Hc);
  k_cvt<<<512, 256, 0, stream>>>(w_hh1, whh1_16, (long)G3 * Hc);
  k_cvt<<<512, 256, 0, stream>>>(w_s, ws16, (long)Hc * Hc);
  k_pad_wr<<<64, 256, 0, stream>>>(w_r, wrp16);
  k_init<<<256, 256, 0, stream>>>(s, sp16, spf, bar);

  k_layer0<<<(2048 * 64) / 8, 256, 0, stream>>>(a16, wih0_16, b_ih0, b_hh0, h0_16);
  k_xp1<<<(2048 * 192) / 8, 256, 0, stream>>>(h0_16, wih1_16, b_ih1, xp1_16);

  float* r_out = (float*)d_out;                      // [B,T,1]
  float* s_out = (float*)d_out + (size_t)Bc * Tc;    // [B,T,H]
  k_scan<<<NWG, 256, 0, stream>>>(xp1_16, whh1_16, b_hh1, ws16, b_s, wrp16, b_r,
                                  sp16, spf, h1_16, r_out, s_out, bar);
}
